// LSTM_Decoder_87995289960806
// MI455X (gfx1250) — matrix operand order repaired and checked by reading, not compiled, not run on hardware
//
#include <hip/hip_runtime.h>
#include <hip/hip_bf16.h>

typedef __bf16 bf16_t;
typedef __attribute__((ext_vector_type(16))) __bf16 v16bf;
typedef __attribute__((ext_vector_type(8)))  __bf16 v8bf;
typedef __attribute__((ext_vector_type(8)))  float  v8f;
typedef __attribute__((ext_vector_type(4)))  float  v4f;

#define Bv   64
#define Sv   512
#define Dv   512
#define Hv   512
#define Lv   2
#define G4H  2048
#define WPG  16      // workgroups cooperating per batch tile
#define NGRP 4       // batch tiles of 16 rows
#define NWG  (WPG*NGRP)
#define NTHREADS 256

// ---- workspace layout (bytes) ----
#define OFF_WIH   0
#define SZ_WIH    ((size_t)Lv*G4H*Dv*2)          // bf16 W_ih  (4 MB)
#define OFF_WHH   (OFF_WIH + SZ_WIH)
#define SZ_WHH    ((size_t)Lv*G4H*Hv*2)          // bf16 W_hh  (4 MB)
#define OFF_HBUF  (OFF_WHH + SZ_WHH)
#define SZ_HBUF   ((size_t)2*Lv*Bv*Hv*2)         // double-buffered bf16 h
#define OFF_CNT   (OFF_HBUF + SZ_HBUF)
#define SZ_CNT    ((size_t)NGRP*2*Sv*4)          // barrier counters

// ---- LDS layout ----
#define LDS_WHH_BYTES  ((size_t)Lv*128*Hv*2)     // 262144: resident W_hh slice
#define LDS_ACT_BYTES  ((size_t)2*16*Hv*2)       //  32768: staged act tiles
#define LDS_GATE_BYTES (8*16*16*4)               //   8192: gate tile exchange
#define LDS_BIAS_BYTES (Lv*4*32*4)               //   1024: combined bias slice
#define LDS_C_BYTES    (Lv*16*32*4)              //   4096: resident c state
#define LDS_TOTAL (LDS_WHH_BYTES + LDS_ACT_BYTES + LDS_GATE_BYTES + LDS_BIAS_BYTES + LDS_C_BYTES)

__device__ __forceinline__ v16bf ld_frag_bf16(const bf16_t* p0, const bf16_t* p1) {
  v8bf a = *(const v8bf*)p0;
  v8bf b = *(const v8bf*)p1;
  v16bf r;
#pragma unroll
  for (int i = 0; i < 8; ++i) { r[i] = a[i]; r[i + 8] = b[i]; }
  return r;
}

// fast sigmoid / tanh on the serial path (hardware rcp + exp)
__device__ __forceinline__ float fast_sigmoid(float x) {
  return __builtin_amdgcn_rcpf(1.0f + __expf(-x));
}
__device__ __forceinline__ float fast_tanh(float x) {
  return 1.0f - 2.0f * __builtin_amdgcn_rcpf(__expf(2.0f * x) + 1.0f);
}

// ---------------- one-time prep: bf16 weights + h state init ----------------
__global__ void lstm_prep_kernel(const float* __restrict__ Wih,
                                 const float* __restrict__ Whh,
                                 const float* __restrict__ ench,
                                 bf16_t* __restrict__ wihb,
                                 bf16_t* __restrict__ whhb,
                                 bf16_t* __restrict__ hbuf) {
  const size_t NW = (size_t)Lv * G4H * Dv;   // 2,097,152 per weight tensor
  const size_t NH = (size_t)Lv * Bv * Hv;    // 65536 (buffer 0, both layers)
  const size_t total  = 2 * NW + NH;
  const size_t stride = (size_t)gridDim.x * blockDim.x;
  for (size_t i = (size_t)blockIdx.x * blockDim.x + threadIdx.x; i < total; i += stride) {
    if (i < NW) {
      wihb[i] = (bf16_t)Wih[i];
    } else if (i < 2 * NW) {
      size_t j = i - NW;
      whhb[j] = (bf16_t)Whh[j];
    } else {
      size_t j = i - 2 * NW;
      size_t r = j & ((size_t)Bv * Hv - 1);
      hbuf[j] = (bf16_t)ench[r];
    }
  }
}

// ---------------- persistent sequential LSTM ----------------
extern __shared__ char smem_raw[];

__global__ __launch_bounds__(NTHREADS, 1) void lstm_seq_kernel(
    const float*  __restrict__ x,      // [B][S][D] fp32
    const bf16_t* __restrict__ wih,    // [L][4H][D] bf16
    const bf16_t* __restrict__ whh_g,  // [L][4H][H] bf16
    const float*  __restrict__ bih,    // [L][4H] fp32 (original)
    const float*  __restrict__ bhh,    // [L][4H] fp32 (original)
    const float*  __restrict__ encc,   // [B][H] fp32
    bf16_t* __restrict__ hbuf,         // [2][L][B][H] bf16 double buffer
    unsigned* __restrict__ cnt,        // [NGRP][2*S] barrier counters
    float* __restrict__ out,           // [B][S][H]
    float* __restrict__ outh,          // [B][H]
    float* __restrict__ outc)          // [B][H]
{
  const int tid  = threadIdx.x;
  const int lane = tid & 31;
  const int wave = tid >> 5;           // 0..7 -> one 16x16 N-tile each
  const int bidx = blockIdx.x;
  const int grp  = bidx >> 4;          // batch tile (16 rows)
  const int wgi  = bidx & 15;          // h-column slice within group
  const int c0   = wgi * 32;           // 32 h-columns per WG

  bf16_t* whh_lds  = (bf16_t*)smem_raw;                                  // [L][128][512]
  bf16_t* act_in   = (bf16_t*)(smem_raw + LDS_WHH_BYTES);                // [16][512]
  bf16_t* act_h    = act_in + 16 * Hv;                                   // [16][512]
  float*  gbuf     = (float*)(smem_raw + LDS_WHH_BYTES + LDS_ACT_BYTES); // [8][16][16]
  float*  bias_lds = gbuf + 8 * 16 * 16;                                 // [L][4][32]
  float*  c_lds    = bias_lds + Lv * 4 * 32;                             // [L][16][32]

  // --- one-time init: W_hh slice to LDS, bias slice, resident c state ---
  for (int i = tid; i < (Lv * 128 * Hv) / 8; i += NTHREADS) {
    int e     = i * 8;
    int layer = e >> 16;               // 128*512 = 65536 elems per layer
    int rem   = e & 65535;
    int lr    = rem >> 9;              // local row 0..127
    int k     = rem & 511;
    int tw    = lr >> 4, tr = lr & 15;
    int tg    = tw >> 1, th = tw & 1;
    int grow  = tg * 512 + c0 + th * 16 + tr;  // global gate row
    const v8bf* src = (const v8bf*)(whh_g + ((size_t)layer * G4H + grow) * Hv + k);
    *(v8bf*)(whh_lds + e) = *src;
  }
  {
    // combined bias: [L][4 gates][32 cols] for this WG's slice
    int layer = tid >> 7, rem = tid & 127;
    int g = rem >> 5, col = rem & 31;
    int bi = layer * G4H + g * 512 + c0 + col;
    bias_lds[tid] = bih[bi] + bhh[bi];
  }
  for (int i = tid; i < Lv * 16 * 32; i += NTHREADS) {
    int m = (i >> 5) & 15, col = i & 31;
    c_lds[i] = encc[(size_t)(grp * 16 + m) * Hv + c0 + col];
  }
  __syncthreads();

  // --- per-wave fragment geometry (16x16x32 bf16, contiguous k = 16*hi + s in A and B) ---
  const int w    = wave;
  const int gate = w >> 1;
  const int half = w & 1;
  const int n15  = lane & 15;
  const int hi   = lane >> 4;                         // lane group 0/1
  const int nrow = gate * 512 + c0 + half * 16 + n15; // B: W row (gate col)
  const int kb_b = hi * 16;                           // B: K base (16 contiguous)
  const int kb_a = hi * 16;                           // A: K base (16 contiguous)

  const bf16_t* whh_base  = whh_lds + (w * 16 + n15) * Hv + kb_b;
  const bf16_t* a_in_lane = act_in + n15 * Hv + kb_a;
  const bf16_t* a_h_lane  = act_h  + n15 * Hv + kb_a;

  int cur = 0;
  for (int s = 0; s < Sv; ++s) {
    const int nxt = cur ^ 1;
#pragma unroll 1
    for (int layer = 0; layer < Lv; ++layer) {
      // ---- stage activation tiles into LDS (uniform, outside K-loop) ----
      if (layer == 0) {
        // x tile fp32 -> bf16
        for (int i = tid * 8; i < 16 * Hv; i += NTHREADS * 8) {
          int m = i >> 9, k = i & 511;
          const float* src = x + ((size_t)(grp * 16 + m) * Sv + s) * Dv + k;
          v4f f0 = *(const v4f*)src;
          v4f f1 = *(const v4f*)(src + 4);
          v8bf o;
#pragma unroll
          for (int j = 0; j < 4; ++j) { o[j] = (bf16_t)f0[j]; o[j + 4] = (bf16_t)f1[j]; }
          *(v8bf*)(act_in + i) = o;
        }
        // prefetch next step's x tile (hides the HBM stream behind compute)
        if (s + 1 < Sv) {
          const float* pf = x + ((size_t)(grp * 16 + (tid >> 4)) * Sv + (s + 1)) * Dv
                              + (tid & 15) * 32;
          __builtin_prefetch(pf, 0, 1);
        }
      } else {
        // layer-1 input = h1(s), already bf16
        for (int i = tid * 8; i < 16 * Hv; i += NTHREADS * 8) {
          int m = i >> 9, k = i & 511;
          const v8bf* src = (const v8bf*)(hbuf +
              (((size_t)nxt * Lv + 0) * Bv + grp * 16 + m) * Hv + k);
          *(v8bf*)(act_in + i) = *src;
        }
      }
      // recurrent h_prev tile
      for (int i = tid * 8; i < 16 * Hv; i += NTHREADS * 8) {
        int m = i >> 9, k = i & 511;
        const v8bf* src = (const v8bf*)(hbuf +
            (((size_t)cur * Lv + layer) * Bv + grp * 16 + m) * Hv + k);
        *(v8bf*)(act_h + i) = *src;
      }
      __syncthreads();

      // ---- branch-free GEMM: A & W_hh from LDS, W_ih streamed from L2 ----
      const bf16_t* bih_ptr = wih + ((size_t)(layer * G4H + nrow)) * Dv + kb_b;
      const bf16_t* bhh_ptr = whh_base + layer * (128 * Hv);

      v8f acc = {0.f, 0.f, 0.f, 0.f, 0.f, 0.f, 0.f, 0.f};
#pragma unroll 4
      for (int k0 = 0; k0 < Dv; k0 += 32) {
        v16bf a_in = ld_frag_bf16(a_in_lane + k0, a_in_lane + k0 + 8);
        v16bf b_in = ld_frag_bf16(bih_ptr + k0, bih_ptr + k0 + 8);
        acc = __builtin_amdgcn_wmma_f32_16x16x32_bf16(false, a_in, false, b_in,
                                                      (short)0, acc, false, false);
        v16bf a_h = ld_frag_bf16(a_h_lane + k0, a_h_lane + k0 + 8);
        v16bf b_h = ld_frag_bf16(bhh_ptr + k0, bhh_ptr + k0 + 8);
        acc = __builtin_amdgcn_wmma_f32_16x16x32_bf16(false, a_h, false, b_h,
                                                      (short)0, acc, false, false);
      }

      // stash gate tile to LDS for i/f/g/o recombination
#pragma unroll
      for (int r = 0; r < 8; ++r) {
        int row = r + 8 * hi;
        gbuf[w * 256 + row * 16 + n15] = acc[r];
      }
      __syncthreads();

      // elementwise LSTM cell: 16 rows x 32 cols per WG (c + bias live in LDS)
      for (int e = tid; e < 512; e += NTHREADS) {
        int M    = e >> 5;
        int col  = e & 31;
        int wi   = col >> 4;
        int n16c = col & 15;
        const float* bl = bias_lds + layer * 128;
        float gi = gbuf[(0 * 2 + wi) * 256 + M * 16 + n16c] + bl[0 * 32 + col];
        float gf = gbuf[(1 * 2 + wi) * 256 + M * 16 + n16c] + bl[1 * 32 + col];
        float gg = gbuf[(2 * 2 + wi) * 256 + M * 16 + n16c] + bl[2 * 32 + col];
        float go = gbuf[(3 * 2 + wi) * 256 + M * 16 + n16c] + bl[3 * 32 + col];
        int   br = grp * 16 + M;
        int   hc = c0 + col;
        int   ci = (layer * 16 + M) * 32 + col;
        float cprev = c_lds[ci];
        float cn = fast_sigmoid(gf) * cprev + fast_sigmoid(gi) * fast_tanh(gg);
        float hn = fast_sigmoid(go) * fast_tanh(cn);
        c_lds[ci] = cn;
        hbuf[(((size_t)nxt * Lv + layer) * Bv + br) * Hv + hc] = (bf16_t)hn;
        if (layer == Lv - 1) {
          out[(size_t)br * Sv * Hv + (size_t)s * Hv + hc] = hn;
          if (s == Sv - 1) {
            outh[br * Hv + hc] = hn;
            outc[br * Hv + hc] = cn;
          }
        }
      }

      // group barrier: 16 WGs sharing this batch tile (monotonic per-phase counter)
      __threadfence();
      __syncthreads();
      if (tid == 0) {
        unsigned* p = cnt + grp * (2 * Sv) + (s * 2 + layer);
        __hip_atomic_fetch_add(p, 1u, __ATOMIC_RELEASE, __HIP_MEMORY_SCOPE_AGENT);
        while (__hip_atomic_load(p, __ATOMIC_ACQUIRE, __HIP_MEMORY_SCOPE_AGENT) < WPG) {
          __builtin_amdgcn_s_sleep(1);
        }
      }
      __syncthreads();
      __threadfence();
    }
    cur ^= 1;
  }
}

extern "C" void kernel_launch(void* const* d_in, const int* in_sizes, int n_in,
                              void* d_out, int out_size, void* d_ws, size_t ws_size,
                              hipStream_t stream) {
  (void)in_sizes; (void)n_in; (void)out_size; (void)ws_size;
  const float* x    = (const float*)d_in[0];
  const float* ench = (const float*)d_in[1];
  const float* encc = (const float*)d_in[2];
  const float* Wih  = (const float*)d_in[3];
  const float* Whh  = (const float*)d_in[4];
  const float* bih  = (const float*)d_in[5];
  const float* bhh  = (const float*)d_in[6];

  char* ws = (char*)d_ws;
  bf16_t*   wihb = (bf16_t*)(ws + OFF_WIH);
  bf16_t*   whhb = (bf16_t*)(ws + OFF_WHH);
  bf16_t*   hbuf = (bf16_t*)(ws + OFF_HBUF);
  unsigned* cnt  = (unsigned*)(ws + OFF_CNT);

  float* out  = (float*)d_out;
  float* outh = out + (size_t)Bv * Sv * Hv;
  float* outc = outh + (size_t)Bv * Hv;

  hipMemsetAsync(cnt, 0, SZ_CNT, stream);
  lstm_prep_kernel<<<2048, 256, 0, stream>>>(Wih, Whh, ench, wihb, whhb, hbuf);
  hipFuncSetAttribute(reinterpret_cast<const void*>(lstm_seq_kernel),
                      hipFuncAttributeMaxDynamicSharedMemorySize, (int)LDS_TOTAL);
  lstm_seq_kernel<<<NWG, NTHREADS, LDS_TOTAL, stream>>>(
      x, wihb, whhb, bih, bhh, encc, hbuf, cnt, out, outh, outc);
}
